// UBlock_64785286693633
// MI455X (gfx1250) — matrix operand order repaired and checked by reading, not compiled, not run on hardware
//
#include <hip/hip_runtime.h>
#include <stdint.h>
#include <stddef.h>

// ---------------------------------------------------------------------------
// 3D V-Net (UBlock) for MI455X / gfx1250.
//  * convs = implicit GEMM on v_wmma_f32_16x16x32_bf16 (wave32 fragments)
//  * activations staged as bf16 NDHWC with channels padded to multiples of 32
//  * per-block weight tiles staged to LDS via global_load_async_to_lds_b128
//    (double-buffered across taps, s_wait_asynccnt + workgroup barrier)
//  * K-chunk count templated (Cp/32 in {1,2,3}) -> unrolled back-to-back WMMA
//  * global_prefetch of next activation row via __builtin_prefetch
//  * BN stats: deterministic two-stage reduction (graph-replay stable)
// ---------------------------------------------------------------------------

typedef __attribute__((ext_vector_type(16))) __bf16       v16bf;
typedef __attribute__((ext_vector_type(8)))  float        v8f;
typedef __attribute__((ext_vector_type(4)))  unsigned int u32x4;

#define BN_EPS 1e-5f
#define NSEG 64     // segments per channel in bn reduction

__device__ __forceinline__ unsigned short f2bf(float f) {
  unsigned int u = __float_as_uint(f);
  unsigned int r = u + 0x7fffu + ((u >> 16) & 1u);   // round-to-nearest-even
  return (unsigned short)(r >> 16);
}

union Frag { v16bf v; u32x4 q[2]; };

// Async global -> LDS copy of one 16B chunk (per enabled lane). ASYNCcnt-tracked.
__device__ __forceinline__ void stage_async(const unsigned short* src,
                                            unsigned short* dstLds) {
  unsigned lds = (unsigned)(uintptr_t)dstLds;   // low 32 bits = LDS byte offset
  asm volatile("global_load_async_to_lds_b128 %0, %1, off"
               :: "v"(lds), "v"(src) : "memory");
}
__device__ __forceinline__ void wait_async0() {
  asm volatile("s_wait_asynccnt 0" ::: "memory");
}

// ---------------------------------------------------------------------------
// Implicit-GEMM conv: act bf16 NDHWC (Cp = KCH*32 padded), wgt [cout][taps][Cp],
// out f32 NCDHW (+coffs channel offset for concat writes).
// grid.y = cout tile  (structurally uniform across the block)
// grid.x = spatial tiles / 8;  one wave = one 16(cout) x 16(x) output tile.
// Per-tap 16xCp weight tile staged once per block into LDS, double-buffered.
// ---------------------------------------------------------------------------
template <int KCH>
__global__ __launch_bounds__(256) void conv_wmma(
    const unsigned short* __restrict__ act,
    const unsigned short* __restrict__ wgt,
    float* __restrict__ out,
    int Din, int Hin, int Win,
    int Dout, int Hout, int Wout,
    int KD, int KH, int KW, int stride, int pad,
    int coffs, int addFlag)
{
  constexpr int Cp = KCH * 32;
  __shared__ unsigned short sw[2][16 * Cp];      // 2/4/6 KB double buffer

  const int tid  = threadIdx.x;
  const int lane = tid & 31;
  const int Wt   = (Wout + 15) >> 4;
  const int tilesPerCo = Dout * Hout * Wt;

  const int cot = blockIdx.y;                    // cout tile, block-uniform
  int r = blockIdx.x * 8 + (tid >> 5);
  const bool ok = r < tilesPerCo;
  int z   = r / (Hout * Wt); r -= z * Hout * Wt;
  int y   = r / Wt;
  int x0  = (r - y * Wt) << 4;

  const int m  = lane & 15;      // A row / C N-lane
  const int hi = lane >> 4;      // K-half selector
  const int xo = x0 + m;         // this lane's output x (B/C N index)

  const int taps = KD * KH * KW;

  // Weight staging: nch = 2*Cp 16B-chunks (64/128/192 => whole waves).
  constexpr int cpr = Cp >> 3;                   // 16B chunks per row
  constexpr int nch = cpr * 16;
  const bool stager = (tid < nch);
  const int  srow = stager ? (tid / cpr) : 0;
  const int  scol = stager ? (tid - srow * cpr) : 0;
  const unsigned short* wbase =
      wgt + (size_t)(cot * 16 + srow) * taps * Cp + (size_t)scol * 8;

  // Prologue: stage tap 0 into buffer 0.
  if (stager) stage_async(wbase, &sw[0][srow * Cp + scol * 8]);

  v8f c = {};
  for (int t = 0; t < taps; ++t) {
    const int cur = t & 1;
    wait_async0();        // this wave's outstanding stage complete
    __syncthreads();      // all waves' chunks visible in LDS
    if (t + 1 < taps && stager)
      stage_async(wbase + (size_t)(t + 1) * Cp,
                  &sw[cur ^ 1][srow * Cp + scol * 8]);

    int fd = t / (KH * KW);
    int rr = t - fd * KH * KW;
    int fh = rr / KW;
    int fw = rr - fh * KW;
    int zz = z * stride + fd - pad;
    int yy = y * stride + fh - pad;
    if ((unsigned)zz < (unsigned)Din && (unsigned)yy < (unsigned)Hin) {
      int xx = xo * stride + fw - pad;
      bool bvalid = (unsigned)xx < (unsigned)Win;
      size_t brow = ((size_t)(zz * Hin + yy) * Win + (size_t)(bvalid ? xx : 0)) * Cp;
      const unsigned short* bp = act + brow + hi * 16;
      // Prefetch next H-row of activations (used by tap t+KW).
      if (yy + 1 < Hin)
        __builtin_prefetch((const void*)(bp + (size_t)Win * Cp), 0, 0);
      u32x4 zq = {0u, 0u, 0u, 0u};
#pragma unroll
      for (int kc = 0; kc < KCH; ++kc) {
        Frag a, b;
        // A from LDS: row m, K = hi*16 + (0..15) (ds_load_b128 x2)
        const unsigned short* ap = &sw[cur][m * Cp + kc * 32 + hi * 16];
        a.q[0] = *(const u32x4*)(ap);
        a.q[1] = *(const u32x4*)(ap + 8);
        // B from global: N = m, K = hi*16 + (0..15) (32B contiguous NDHWC)
        b.q[0] = bvalid ? *(const u32x4*)(bp + kc * 32)     : zq;
        b.q[1] = bvalid ? *(const u32x4*)(bp + kc * 32 + 8) : zq;
        c = __builtin_amdgcn_wmma_f32_16x16x32_bf16(false, a.v, false, b.v,
                                                    (short)0, c, false, false);
      }
    }
    __syncthreads();      // everyone done with sw[cur] before it is restaged
  }

  if (ok && xo < Wout) {
    size_t cs   = (size_t)Dout * Hout * Wout;
    size_t base = (size_t)(coffs + cot * 16) * cs +
                  ((size_t)z * Hout + y) * Wout + xo;
#pragma unroll
    for (int v = 0; v < 8; ++v) {
      int mm = v + hi * 8;
      float val = c[v];
      float* p = out + base + (size_t)mm * cs;
      if (addFlag) val += *p;
      *p = val;
    }
  }
}

// ---------------------------------------------------------------------------
// Transposed conv k=2 s=2 VALID as per-parity GEMM: out[2zi+kd][2yi+kh][2xi+kw]
// gets tap (kd,kh,kw). Tap varies per wave within a block, so weights are read
// directly from global (tiny, L0-resident). grid.y = cout tile.
// ---------------------------------------------------------------------------
template <int KCH>
__global__ __launch_bounds__(256) void deconv_wmma(
    const unsigned short* __restrict__ act,   // bf16 NDHWC, spatial S, Cp
    const unsigned short* __restrict__ wgt,   // [cout][8][Cp]
    float* __restrict__ out,                  // f32 NCDHW, spatial 2S
    int S, int coffs)
{
  constexpr int Cp = KCH * 32;
  const int lane = threadIdx.x & 31;
  const int S2 = 2 * S;
  const int Kt = (S + 15) >> 4;
  const int tilesPerCo = S2 * S2 * 2 * Kt;

  const int cot = blockIdx.y;
  int r = blockIdx.x * 8 + (threadIdx.x >> 5);
  if (r >= tilesPerCo) return;
  int z   = r / (S2 * 2 * Kt); r -= z * S2 * 2 * Kt;
  int y   = r / (2 * Kt);      r -= y * (2 * Kt);
  int p   = r / Kt;
  int t   = r - p * Kt;

  const int m  = lane & 15;
  const int hi = lane >> 4;
  const int zin = z >> 1, yin = y >> 1;
  const int tap = ((z & 1) * 2 + (y & 1)) * 2 + p;
  const int xin = t * 16 + m;
  const bool valid = xin < S;

  size_t brow = ((size_t)(zin * S + yin) * S + (size_t)(valid ? xin : 0)) * Cp;
  const unsigned short* wrow = wgt + ((size_t)(cot * 16 + m) * 8 + tap) * Cp;
  const unsigned short* bp = act + brow + hi * 16;
  u32x4 zq = {0u, 0u, 0u, 0u};

  v8f c = {};
#pragma unroll
  for (int kc = 0; kc < KCH; ++kc) {
    Frag a, b;
    const unsigned short* ap = wrow + kc * 32 + hi * 16;
    a.q[0] = *(const u32x4*)(ap);
    a.q[1] = *(const u32x4*)(ap + 8);
    b.q[0] = valid ? *(const u32x4*)(bp + kc * 32)     : zq;
    b.q[1] = valid ? *(const u32x4*)(bp + kc * 32 + 8) : zq;
    c = __builtin_amdgcn_wmma_f32_16x16x32_bf16(false, a.v, false, b.v,
                                                (short)0, c, false, false);
  }

  if (valid) {
    int xo = p + 2 * xin;
    size_t cs   = (size_t)S2 * S2 * S2;
    size_t base = (size_t)(coffs + cot * 16) * cs + ((size_t)z * S2 + y) * S2 + xo;
#pragma unroll
    for (int v = 0; v < 8; ++v)
      out[base + (size_t)(v + hi * 8) * cs] = c[v];
  }
}

// ---------------------------------------------------------------------------
// BN statistics: deterministic two-stage (partial sums -> finalize).
// ---------------------------------------------------------------------------
__global__ __launch_bounds__(256) void bn_partial(const float* __restrict__ x,
                                                  float* __restrict__ part, int S3)
{
  int c = blockIdx.x / NSEG, seg = blockIdx.x % NSEG;
  int chunk = (S3 + NSEG - 1) / NSEG;
  int lo = seg * chunk;
  int hiB = lo + chunk; if (hiB > S3) hiB = S3;
  const float* p = x + (size_t)c * S3;
  float s = 0.f, s2 = 0.f;
  for (int i = lo + threadIdx.x; i < hiB; i += blockDim.x) {
    float v = p[i]; s += v; s2 += v * v;
  }
  __shared__ float r1[256], r2[256];
  r1[threadIdx.x] = s; r2[threadIdx.x] = s2;
  __syncthreads();
  for (int o = 128; o; o >>= 1) {
    if ((int)threadIdx.x < o) {
      r1[threadIdx.x] += r1[threadIdx.x + o];
      r2[threadIdx.x] += r2[threadIdx.x + o];
    }
    __syncthreads();
  }
  if (threadIdx.x == 0) {
    part[(c * NSEG + seg) * 2 + 0] = r1[0];
    part[(c * NSEG + seg) * 2 + 1] = r2[0];
  }
}

__global__ void bn_final(const float* __restrict__ part, float* __restrict__ mv,
                         int Cc, int S3)
{
  int c = blockIdx.x * blockDim.x + threadIdx.x;
  if (c >= Cc) return;
  float s = 0.f, s2 = 0.f;
  for (int i = 0; i < NSEG; ++i) {
    s  += part[(c * NSEG + i) * 2 + 0];
    s2 += part[(c * NSEG + i) * 2 + 1];
  }
  float mu = s / (float)S3;
  mv[2 * c + 0] = mu;
  mv[2 * c + 1] = s2 / (float)S3 - mu * mu;
}

// BN+ReLU (mode=1) or plain convert (mode=0): f32 NCDHW -> bf16 NDHWC padded.
__global__ void bnrelu_bf16(const float* __restrict__ x, const float* __restrict__ mv,
                            const float* __restrict__ g, const float* __restrict__ b,
                            unsigned short* __restrict__ o,
                            int Cc, int Cp, int S3, int mode)
{
  size_t n = (size_t)S3 * Cp;
  for (size_t i = (size_t)blockIdx.x * blockDim.x + threadIdx.x; i < n;
       i += (size_t)gridDim.x * blockDim.x) {
    int c = (int)(i % (size_t)Cp);
    size_t s = i / (size_t)Cp;
    float v = 0.f;
    if (c < Cc) {
      v = x[(size_t)c * S3 + s];
      if (mode) {
        float mu = mv[2 * c], var = mv[2 * c + 1];
        v = (v - mu) * rsqrtf(var + BN_EPS) * g[c] + b[c];
        v = v > 0.f ? v : 0.f;
      }
    }
    o[i] = f2bf(v);
  }
}

// f32 OIDHW weights -> bf16 [cout][taps][Cp] (channel-contiguous K, zero pad).
__global__ void wconv_bf16(const float* __restrict__ w, unsigned short* __restrict__ o,
                           int cout, int cin, int taps, int Cp)
{
  int n = cout * taps * Cp;
  for (int i = blockIdx.x * blockDim.x + threadIdx.x; i < n;
       i += gridDim.x * blockDim.x) {
    int ci = i % Cp;
    int t  = (i / Cp) % taps;
    int co = i / (Cp * taps);
    float v = (ci < cin) ? w[((size_t)co * cin + ci) * taps + t] : 0.f;
    o[i] = f2bf(v);
  }
}

__global__ void addk(float* __restrict__ o, const float* __restrict__ a, size_t n) {
  for (size_t i = (size_t)blockIdx.x * blockDim.x + threadIdx.x; i < n;
       i += (size_t)gridDim.x * blockDim.x) o[i] += a[i];
}
__global__ void copyk(float* __restrict__ o, const float* __restrict__ a, size_t n) {
  for (size_t i = (size_t)blockIdx.x * blockDim.x + threadIdx.x; i < n;
       i += (size_t)gridDim.x * blockDim.x) o[i] = a[i];
}

// ---------------------------------------------------------------------------
// Host orchestration
// ---------------------------------------------------------------------------
static const int SS[4]   = {96, 48, 24, 12};
static const int INC[4]  = {16, 32, 48, 64};
static const int OUTC[4] = {16, 32, 48, 64};

static inline int cp32(int c) { return (c + 31) & ~31; }

struct ResP { const float *W1, *W2, *Wskip, *b1, *b2, *g1, *g2; int cin, cout; };
struct LvP  { ResP blk; const float *cW, *cB, *cG, *dW, *dB, *dG; ResP tail; };

struct Ctx {
  hipStream_t st;
  char*  base;
  size_t cap, off;
  void* alloc(size_t bytes) {
    size_t a = (off + 255) & ~(size_t)255;
    if (a + bytes > cap) a = 0;          // best-effort wrap if ws is small
    off = a + bytes;
    return base + a;
  }
};

static inline int egrid(size_t n) {
  size_t g = (n + 255) / 256;
  if (g > (size_t)262144) g = 262144;
  if (g < 1) g = 1;
  return (int)g;
}

static void run_bn(Ctx& C, const float* x, const float* g, const float* b,
                   unsigned short* t, int Cc, int Cp, int S3, int mode)
{
  size_t mk = C.off;
  float* part = (float*)C.alloc((size_t)Cc * NSEG * 2 * sizeof(float));
  float* mv   = (float*)C.alloc((size_t)Cc * 2 * sizeof(float));
  if (mode) {
    bn_partial<<<Cc * NSEG, 256, 0, C.st>>>(x, part, S3);
    bn_final<<<(Cc + 63) / 64, 64, 0, C.st>>>(part, mv, Cc, S3);
  }
  bnrelu_bf16<<<egrid((size_t)S3 * Cp), 256, 0, C.st>>>(x, mv, g, b, t, Cc, Cp, S3, mode);
  C.off = mk;
}

static void run_wconv(Ctx& C, const float* w, unsigned short* o,
                      int cout, int cin, int taps, int Cp)
{
  wconv_bf16<<<egrid((size_t)cout * taps * Cp), 256, 0, C.st>>>(w, o, cout, cin, taps, Cp);
}

static void run_conv(Ctx& C, const unsigned short* act, const unsigned short* w,
                     float* out, int Cp, int Sin, int cout, int Sout,
                     int K, int stride, int pad, int coffs, int add)
{
  int Wt = (Sout + 15) >> 4;
  int tilesPerCo = Sout * Sout * Wt;            // divisible by 8 for all shapes
  dim3 grid((tilesPerCo + 7) / 8, cout >> 4);
  switch (Cp >> 5) {
    case 1: conv_wmma<1><<<grid, 256, 0, C.st>>>(act, w, out, Sin, Sin, Sin,
                Sout, Sout, Sout, K, K, K, stride, pad, coffs, add); break;
    case 2: conv_wmma<2><<<grid, 256, 0, C.st>>>(act, w, out, Sin, Sin, Sin,
                Sout, Sout, Sout, K, K, K, stride, pad, coffs, add); break;
    default: conv_wmma<3><<<grid, 256, 0, C.st>>>(act, w, out, Sin, Sin, Sin,
                Sout, Sout, Sout, K, K, K, stride, pad, coffs, add); break;
  }
}

static void run_deconv(Ctx& C, const unsigned short* act, const unsigned short* w,
                       float* out, int Cp, int S, int cout, int coffs)
{
  int S2 = 2 * S;
  int Kt = (S + 15) >> 4;
  int tilesPerCo = S2 * S2 * 2 * Kt;
  dim3 grid((tilesPerCo + 7) / 8, cout >> 4);
  switch (Cp >> 5) {
    case 1: deconv_wmma<1><<<grid, 256, 0, C.st>>>(act, w, out, S, coffs); break;
    case 2: deconv_wmma<2><<<grid, 256, 0, C.st>>>(act, w, out, S, coffs); break;
    default: deconv_wmma<3><<<grid, 256, 0, C.st>>>(act, w, out, S, coffs); break;
  }
}

static void run_res(Ctx& C, const float* x, const ResP& p, float* out, int S)
{
  size_t mk = C.off;
  int S3 = S * S * S;
  int Cpi = cp32(p.cin), Cpo = cp32(p.cout);
  unsigned short* t1 = (unsigned short*)C.alloc((size_t)Cpi * S3 * 2);
  unsigned short* w1 = (unsigned short*)C.alloc((size_t)p.cout * 27 * Cpi * 2);
  float*          h1 = (float*)C.alloc((size_t)p.cout * S3 * sizeof(float));
  unsigned short* t2 = (unsigned short*)C.alloc((size_t)Cpo * S3 * 2);
  unsigned short* w2 = (unsigned short*)C.alloc((size_t)p.cout * 27 * Cpo * 2);

  run_bn(C, x, p.g1, p.b1, t1, p.cin, Cpi, S3, 1);
  run_wconv(C, p.W1, w1, p.cout, p.cin, 27, Cpi);
  run_conv(C, t1, w1, h1, Cpi, S, p.cout, S, 3, 1, 1, 0, 0);

  run_bn(C, h1, p.g2, p.b2, t2, p.cout, Cpo, S3, 1);
  run_wconv(C, p.W2, w2, p.cout, p.cout, 27, Cpo);
  run_conv(C, t2, w2, out, Cpo, S, p.cout, S, 3, 1, 1, 0, 0);

  if (p.Wskip) {
    unsigned short* wsk = (unsigned short*)C.alloc((size_t)p.cout * Cpi * 2);
    run_bn(C, x, nullptr, nullptr, t1, p.cin, Cpi, S3, 0);   // plain convert
    run_wconv(C, p.Wskip, wsk, p.cout, p.cin, 1, Cpi);
    run_conv(C, t1, wsk, out, Cpi, S, p.cout, S, 1, 1, 0, 0, 1);  // add=1
  } else {
    addk<<<egrid((size_t)p.cout * S3), 256, 0, C.st>>>(out, x, (size_t)p.cout * S3);
  }
  C.off = mk;
}

static void run_ublock(Ctx& C, int L, const float* x, const LvP* Lv,
                       const ResP& inner, float* const* reg)
{
  if (L == 3) { run_res(C, x, inner, reg[3], SS[3]); return; }

  int S = SS[L], S3 = S * S * S, Sh = S >> 1, Sh3 = Sh * Sh * Sh;
  int Ci = INC[L], Co = OUTC[L], Cd = INC[L + 1], Cu = OUTC[L + 1];
  size_t mk = C.off;

  float* ident = (float*)C.alloc((size_t)Ci * S3 * sizeof(float));
  run_res(C, x, Lv[L].blk, ident, S);

  // down path: bn_relu -> conv k2 s2 VALID
  float* y = (float*)C.alloc((size_t)Cd * Sh3 * sizeof(float));
  {
    size_t mk2 = C.off;
    int Cp = cp32(Ci);
    unsigned short* t  = (unsigned short*)C.alloc((size_t)Cp * S3 * 2);
    unsigned short* wb = (unsigned short*)C.alloc((size_t)Cd * 8 * Cp * 2);
    run_bn(C, ident, Lv[L].cG, Lv[L].cB, t, Ci, Cp, S3, 1);
    run_wconv(C, Lv[L].cW, wb, Cd, Ci, 8, Cp);
    run_conv(C, t, wb, y, Cp, S, Cd, Sh, 2, 2, 0, 0, 0);
    C.off = mk2;
  }

  run_ublock(C, L + 1, y, Lv, inner, reg);

  // up path: bn_relu -> deconv into concat[Ci:], identity -> concat[0:Ci]
  float* concat = (float*)C.alloc((size_t)(Ci + Co) * S3 * sizeof(float));
  {
    size_t mk2 = C.off;
    int Cp = cp32(Cu);
    unsigned short* t  = (unsigned short*)C.alloc((size_t)Cp * Sh3 * 2);
    unsigned short* wb = (unsigned short*)C.alloc((size_t)Co * 8 * Cp * 2);
    run_bn(C, reg[L + 1], Lv[L].dG, Lv[L].dB, t, Cu, Cp, Sh3, 1);
    run_wconv(C, Lv[L].dW, wb, Co, Cu, 8, Cp);
    copyk<<<egrid((size_t)Ci * S3), 256, 0, C.st>>>(concat, ident, (size_t)Ci * S3);
    run_deconv(C, t, wb, concat, Cp, Sh, Co, Ci);
    C.off = mk2;
  }

  run_res(C, concat, Lv[L].tail, reg[L], S);
  C.off = mk;
}

extern "C" void kernel_launch(void* const* d_in, const int* in_sizes, int n_in,
                              void* d_out, int out_size, void* d_ws, size_t ws_size,
                              hipStream_t stream)
{
  (void)in_sizes; (void)out_size;
  const float* x = (const float*)d_in[0];

  // Params assumed flattened as JAX tree leaves (dict keys sorted):
  // per level: blocks[0]{W1,W2,b1,b2,g1,g2}, conv_W, conv_b, conv_g,
  //            deconv_W, deconv_b, deconv_g, tail[0]{W1,W2,Wskip,b1,b2,g1,g2}, u...
  int cur = 1;
  auto nx = [&]() -> const float* {
    int i = (cur < n_in) ? cur : (n_in - 1);
    ++cur;
    return (const float*)d_in[i];
  };

  LvP Lv[3]; ResP inner;
  for (int L = 0; L < 3; ++L) {
    Lv[L].blk  = ResP{nx(), nx(), nullptr, nx(), nx(), nx(), nx(), INC[L], INC[L]};
    Lv[L].cW = nx(); Lv[L].cB = nx(); Lv[L].cG = nx();
    Lv[L].dW = nx(); Lv[L].dB = nx(); Lv[L].dG = nx();
    Lv[L].tail = ResP{nx(), nx(), nx(), nx(), nx(), nx(), nx(),
                      OUTC[L] + INC[L], OUTC[L]};
  }
  inner = ResP{nx(), nx(), nullptr, nx(), nx(), nx(), nx(), INC[3], OUTC[3]};

  // Output regions: tuple (out0, out1, out2, out3) flattened in return order.
  float* reg[4];
  {
    float* ob = (float*)d_out;
    size_t o = 0;
    for (int L = 0; L < 4; ++L) {
      reg[L] = ob + o;
      o += (size_t)OUTC[L] * SS[L] * SS[L] * SS[L];
    }
  }

  Ctx C{stream, (char*)d_ws, ws_size, 0};
  run_ublock(C, 0, x, Lv, inner, reg);
}
